// DG_46488726012428
// MI455X (gfx1250) — matrix operand order repaired and checked by reading, not compiled, not run on hardware
//
#include <hip/hip_runtime.h>

typedef __attribute__((ext_vector_type(16))) __bf16 v16bf;
typedef __attribute__((ext_vector_type(8)))  float  v8f;

#define H_UNITS 16384
#define IN_SIZE 2048
#define BATCH   512
#define TOPK    64u
#define DECAY   0.95f

#define MT      4                       // M-tiles per wave
#define KSTEPS  (IN_SIZE / 32)          // 64
#define MTILES  (BATCH / 16)            // 32

// ------------------------------------------------------------------
// Prep: split inputs into bf16 hi/lo planes, pre-swizzled into WMMA
// A-fragment order: frag[(tileM*KSTEPS + ks)*32 + lane][16 bf16].
// A 16-bit 16x32 fragment (ISA 7.12.2): lane L holds row M=L%16;
// VGPR i -> K = (i/4)*16 + (L<16?0:8) + (i%4)*2 (+e).
// ------------------------------------------------------------------
__global__ __launch_bounds__(256) void dg_prep_a(const float* __restrict__ A,
                                                 __bf16* __restrict__ Ahi,
                                                 __bf16* __restrict__ Alo)
{
    const int t     = blockIdx.x * blockDim.x + threadIdx.x;  // 0..65535
    const int lane  = t & 31;
    const int ks    = (t >> 5) & (KSTEPS - 1);
    const int tileM = t >> 11;                                // 0..31

    const int rowA = tileM * 16 + (lane & 15);
    const int kA   = (lane >> 4) * 8;
    const size_t fbase = ((size_t)(tileM * KSTEPS + ks) * 32 + lane) * 16;

    __bf16 hi[16], lo[16];
#pragma unroll
    for (int i = 0; i < 8; ++i) {
        const int k = ks * 32 + (i >> 2) * 16 + kA + (i & 3) * 2;
        const float2 p = *(const float2*)(A + (size_t)rowA * IN_SIZE + k);
        __bf16 h0 = (__bf16)p.x, h1 = (__bf16)p.y;
        hi[2 * i]     = h0;
        hi[2 * i + 1] = h1;
        lo[2 * i]     = (__bf16)(p.x - (float)h0);
        lo[2 * i + 1] = (__bf16)(p.y - (float)h1);
    }
#pragma unroll
    for (int e = 0; e < 16; ++e) { Ahi[fbase + e] = hi[e]; Alo[fbase + e] = lo[e]; }
}

// ------------------------------------------------------------------
// GEMM: enc[m][n] = sum_k inputs[m][k] * W[n][k]
// bf16x3 split precision, f32 accumulate. A fragments precomputed;
// W split inline (L2-resident, co-executes with WMMA).
// ------------------------------------------------------------------
__global__ __launch_bounds__(256) void dg_gemm(const __bf16* __restrict__ Ahi,
                                               const __bf16* __restrict__ Alo,
                                               const float* __restrict__ W,
                                               float* __restrict__ C)
{
    const int lane   = threadIdx.x & 31;
    const int wave   = (int)((blockIdx.x * blockDim.x + threadIdx.x) >> 5);
    const int mGrps  = MTILES / MT;                // 8
    const int tileN  = wave / mGrps;               // 0..1023
    const int mBase  = (wave % mGrps) * MT;        // tileM varies fastest across waves
    const int l15    = lane & 15;
    const int hiHalf = lane >> 4;

    const int rowW = tileN * 16 + l15;             // B column n = W row
    const int kB   = hiHalf * 8;                   // lanes 16-31 take K+8
    const float* Wrow = W + (size_t)rowW * IN_SIZE;

    v8f acc[MT];
#pragma unroll
    for (int t = 0; t < MT; ++t)
#pragma unroll
        for (int r = 0; r < 8; ++r) acc[t][r] = 0.0f;

    for (int ks = 0; ks < KSTEPS; ++ks) {
        const int k0 = ks * 32;

        // ---- B tile: per-lane K slice is two runs of 8 floats (kB, 16+kB) -> 4x b128 ----
        const float4 q0 = *(const float4*)(Wrow + k0 + kB + 0);
        const float4 q1 = *(const float4*)(Wrow + k0 + kB + 4);
        const float4 q2 = *(const float4*)(Wrow + k0 + kB + 16);
        const float4 q3 = *(const float4*)(Wrow + k0 + kB + 20);
        float bf[16] = {q0.x, q0.y, q0.z, q0.w, q1.x, q1.y, q1.z, q1.w,
                        q2.x, q2.y, q2.z, q2.w, q3.x, q3.y, q3.z, q3.w};
        v16bf b_hi, b_lo;
#pragma unroll
        for (int e = 0; e < 16; ++e) {
            __bf16 h = (__bf16)bf[e];
            b_hi[e] = h;
            b_lo[e] = (__bf16)(bf[e] - (float)h);
        }
        if (k0 + 256 < IN_SIZE)
            __builtin_prefetch(Wrow + k0 + 256, 0, 1);

#pragma unroll
        for (int t = 0; t < MT; ++t) {
            const size_t fbase = ((size_t)((mBase + t) * KSTEPS + ks) * 32 + lane) * 16;
            const v16bf a_hi = *(const v16bf*)(Ahi + fbase);
            const v16bf a_lo = *(const v16bf*)(Alo + fbase);
            acc[t] = __builtin_amdgcn_wmma_f32_16x16x32_bf16(false, a_hi, false, b_hi,
                                                             (short)0, acc[t], false, false);
            acc[t] = __builtin_amdgcn_wmma_f32_16x16x32_bf16(false, a_lo, false, b_hi,
                                                             (short)0, acc[t], false, false);
            acc[t] = __builtin_amdgcn_wmma_f32_16x16x32_bf16(false, a_hi, false, b_lo,
                                                             (short)0, acc[t], false, false);
        }
    }

    // C/D layout: VGPR r -> M = r + hiHalf*8, N = l15
    const int nOut = tileN * 16 + l15;
#pragma unroll
    for (int t = 0; t < MT; ++t)
#pragma unroll
        for (int r = 0; r < 8; ++r) {
            const int m = (mBase + t) * 16 + r + hiHalf * 8;
            C[(size_t)m * H_UNITS + nOut] = acc[t][r];
        }
}

// ------------------------------------------------------------------
// Sortable key: order-preserving f32 -> u32 map (handles negatives)
// ------------------------------------------------------------------
__device__ __forceinline__ unsigned f2sort(float x)
{
    unsigned u = __float_as_uint(x);
    return (u & 0x80000000u) ? ~u : (u | 0x80000000u);
}

// ------------------------------------------------------------------
// Sequential inhibition scan: one persistent workgroup, 512 rows.
// ------------------------------------------------------------------
#define ST 512
#define SE (H_UNITS / ST)   // 32 elements per thread

__global__ __launch_bounds__(ST) void dg_scan(float* __restrict__ enc)
{
    __shared__ unsigned hist[256];
    __shared__ unsigned tiecnt[ST];
    __shared__ unsigned s_prefix, s_remk;

    const int tid  = threadIdx.x;
    const int base = tid * SE;

    float inhib[SE];
#pragma unroll
    for (int i = 0; i < SE; ++i) inhib[i] = 0.0f;

    for (int row = 0; row < BATCH; ++row) {
        float* e = enc + (size_t)row * H_UNITS;
        float    ev[SE];
        unsigned u[SE];
#pragma unroll
        for (int i = 0; i < SE; ++i) {
            ev[i] = e[base + i];
            u[i]  = f2sort(fabsf(ev[i]) * (1.0f - inhib[i]));
        }
        if (tid == 0) { s_prefix = 0u; s_remk = TOPK; }

        for (int pass = 0; pass < 4; ++pass) {
            const int sh = 24 - 8 * pass;
            for (int b = tid; b < 256; b += ST) hist[b] = 0u;
            __syncthreads();
            const unsigned pref  = s_prefix;
            const unsigned hmask = (pass == 0) ? 0u : (0xFFFFFFFFu << (sh + 8));
#pragma unroll
            for (int i = 0; i < SE; ++i)
                if ((u[i] & hmask) == (pref & hmask))
                    atomicAdd(&hist[(u[i] >> sh) & 255u], 1u);
            __syncthreads();
            if (tid == 0) {
                unsigned remk = s_remk, c = 0;
                int b = 255;
                for (; b > 0; --b) { c += hist[b]; if (c >= remk) break; }
                if (c < remk) c += hist[0];
                s_prefix = pref | ((unsigned)b << sh);
                s_remk   = remk - (c - hist[b]);
            }
            __syncthreads();
        }
        const unsigned T    = s_prefix;
        const unsigned remk = s_remk;

        // deterministic lowest-index-first tie break
        unsigned lt = 0;
#pragma unroll
        for (int i = 0; i < SE; ++i) lt += (u[i] == T) ? 1u : 0u;
        tiecnt[tid] = lt;
        __syncthreads();
        if (tid == 0) {
            unsigned run = 0;
            for (int t = 0; t < ST; ++t) { unsigned c = tiecnt[t]; tiecnt[t] = run; run += c; }
        }
        __syncthreads();
        unsigned off = tiecnt[tid];
#pragma unroll
        for (int i = 0; i < SE; ++i) {
            int sel = (u[i] > T) ? 1 : 0;
            if (u[i] == T) { sel = (off < remk) ? 1 : 0; ++off; }
            e[base + i] = sel ? ev[i] : 0.0f;            // filtered, in place
            inhib[i]    = inhib[i] * DECAY + (float)sel;
        }
        __syncthreads();
    }
}

// ------------------------------------------------------------------
// Final per-row top-64 mask over filtered (parallel over rows).
// ------------------------------------------------------------------
__global__ __launch_bounds__(ST) void dg_final(const float* __restrict__ filt,
                                               float* __restrict__ out)
{
    __shared__ unsigned hist[256];
    __shared__ unsigned tiecnt[ST];
    __shared__ unsigned s_prefix, s_remk;

    const int tid  = threadIdx.x;
    const int row  = blockIdx.x;
    const int base = tid * SE;
    const float* f = filt + (size_t)row * H_UNITS;

    unsigned u[SE];
#pragma unroll
    for (int i = 0; i < SE; ++i) u[i] = f2sort(f[base + i]);

    if (tid == 0) { s_prefix = 0u; s_remk = TOPK; }

    for (int pass = 0; pass < 4; ++pass) {
        const int sh = 24 - 8 * pass;
        for (int b = tid; b < 256; b += ST) hist[b] = 0u;
        __syncthreads();
        const unsigned pref  = s_prefix;
        const unsigned hmask = (pass == 0) ? 0u : (0xFFFFFFFFu << (sh + 8));
#pragma unroll
        for (int i = 0; i < SE; ++i)
            if ((u[i] & hmask) == (pref & hmask))
                atomicAdd(&hist[(u[i] >> sh) & 255u], 1u);
        __syncthreads();
        if (tid == 0) {
            unsigned remk = s_remk, c = 0;
            int b = 255;
            for (; b > 0; --b) { c += hist[b]; if (c >= remk) break; }
            if (c < remk) c += hist[0];
            s_prefix = pref | ((unsigned)b << sh);
            s_remk   = remk - (c - hist[b]);
        }
        __syncthreads();
    }
    const unsigned T    = s_prefix;
    const unsigned remk = s_remk;

    unsigned lt = 0;
#pragma unroll
    for (int i = 0; i < SE; ++i) lt += (u[i] == T) ? 1u : 0u;
    tiecnt[tid] = lt;
    __syncthreads();
    if (tid == 0) {
        unsigned run = 0;
        for (int t = 0; t < ST; ++t) { unsigned c = tiecnt[t]; tiecnt[t] = run; run += c; }
    }
    __syncthreads();
    unsigned off = tiecnt[tid];
#pragma unroll
    for (int i = 0; i < SE; ++i) {
        int sel = (u[i] > T) ? 1 : 0;
        if (u[i] == T) { sel = (off < remk) ? 1 : 0; ++off; }
        out[(size_t)row * H_UNITS + base + i] = sel ? 1.0f : 0.0f;
    }
}

// ------------------------------------------------------------------
extern "C" void kernel_launch(void* const* d_in, const int* in_sizes, int n_in,
                              void* d_out, int out_size, void* d_ws, size_t ws_size,
                              hipStream_t stream)
{
    const float* inputs = (const float*)d_in[0];   // [512, 2048]
    const float* W      = (const float*)d_in[1];   // [16384, 2048]
    float* out = (float*)d_out;                    // [512, 16384]

    // workspace layout: enc (BATCH*H_UNITS f32) | Ahi | Alo (BATCH*IN_SIZE bf16 each)
    char* ws = (char*)d_ws;
    float* enc   = (float*)ws;
    __bf16* Ahi = (__bf16*)(ws + (size_t)BATCH * H_UNITS * sizeof(float));
    __bf16* Alo = Ahi + (size_t)BATCH * IN_SIZE;

    // 1) pre-swizzle + split A into WMMA fragment order: 65536 threads
    dg_prep_a<<<(MTILES * KSTEPS * 32) / 256, 256, 0, stream>>>(inputs, Ahi, Alo);

    // 2) GEMM: 1024 N-tiles * 8 M-groups = 8192 waves, 8 waves/block
    const int waves  = (H_UNITS / 16) * (MTILES / MT);
    const int blocks = waves * 32 / 256;
    dg_gemm<<<blocks, 256, 0, stream>>>(Ahi, Alo, W, enc);

    // 3) sequential inhibition scan: one persistent workgroup
    dg_scan<<<1, ST, 0, stream>>>(enc);

    // 4) final per-row top-k mask
    dg_final<<<BATCH, ST, 0, stream>>>(enc, out);
}
